// NPT_16200616640975
// MI455X (gfx1250) — matrix operand order repaired and checked by reading, not compiled, not run on hardware
//
#include <hip/hip_runtime.h>
#include <hip/hip_bf16.h>
#include <math.h>

// ---------------------------------------------------------------------------
// Problem constants (fixed by the reference)
// ---------------------------------------------------------------------------
#define BB   8
#define NN   1024
#define DD   512
#define HH   8
#define DHH  64
#define FFD  2048
#define MM   (BB * NN)          // 8192 rows total

typedef __bf16 bf16_t;
typedef bf16_t v16bf __attribute__((ext_vector_type(16)));
typedef float  v8f   __attribute__((ext_vector_type(8)));
typedef unsigned int tdm_v4u __attribute__((ext_vector_type(4)));
typedef int          tdm_v8i __attribute__((ext_vector_type(8)));
typedef int          tdm_v4i __attribute__((ext_vector_type(4)));

union FragBF {
    v16bf v;
    uint4 u[2];
};

__device__ __forceinline__ v8f wmma_bf16(v16bf a, v16bf b, v8f c) {
    // D(f32,16x16) = A(bf16,16x32) * B(bf16,32x16) + C
    return __builtin_amdgcn_wmma_f32_16x16x32_bf16(
        false, a, false, b, (short)0, c, false, false);
}

__device__ __forceinline__ float gelu_new(float x) {
    return 0.5f * x * (1.0f + tanhf(0.7978845608028654f * (x + 0.044715f * x * x * x)));
}

// ---------------------------------------------------------------------------
// TDM: DMA a 2D bf16 tile [tile_d1 rows x 64 cols] from global (row stride =
// stride_elems) into LDS at lds_off.  LDS padding: +4 dwords every 32 dwords
// => LDS row stride of 72 halfs (bank-conflict-free for the fragment reads).
// D# layout per CDNA5 ISA ch.8 (group0 128b, group1 256b; groups 2/3 unused).
// This toolchain exposes the 6-arg builtin (g0,g1,g2,g3,g4,cpol).
// ---------------------------------------------------------------------------
__device__ __forceinline__ void tdm_load_2d(unsigned int lds_off,
                                            const bf16_t* gptr,
                                            int tile_d1, int stride_elems)
{
    unsigned long long ga = (unsigned long long)(uintptr_t)gptr;
    tdm_v4u g0;
    g0[0] = 1u;                                   // count=1, user descriptor
    g0[1] = lds_off;                              // LDS byte address
    g0[2] = (unsigned int)ga;                     // global_addr[31:0]
    g0[3] = (unsigned int)((ga >> 32) & 0x1FFFFFFu) | 0x80000000u; // addr[56:32] | type=2

    tdm_v8i g1;
    // data_size=2B (code 1) | pad_enable | pad_interval=32dw (code 4) |
    // pad_amount=4dw (code 3); workgroup_mask=0 (not in a cluster)
    g1[0] = (1 << 16) | (1 << 20) | (4 << 22) | (3 << 25);
    const int td0 = 1 << 28;                      // tensor_dim0 (huge: no OOB)
    const int td1 = 1 << 28;                      // tensor_dim1 (huge: no OOB)
    g1[1] = (td0 & 0xFFFF) << 16;                 // [63:48] = tensor_dim0 lo16
    g1[2] = ((td0 >> 16) & 0xFFFF) | ((td1 & 0xFFFF) << 16);
    g1[3] = ((td1 >> 16) & 0xFFFF) | (64 << 16);  // tile_dim0 = 64 elements
    g1[4] = tile_d1 & 0xFFFF;                     // tile_dim1; tile_dim2=0
    g1[5] = stride_elems;                         // tensor_dim0_stride[31:0]
    g1[6] = 0;                                    // stride[47:32], dim1_stride lo
    g1[7] = 0;

    tdm_v4i z4 = {0, 0, 0, 0};
    tdm_v8i z8 = {0, 0, 0, 0, 0, 0, 0, 0};
    __builtin_amdgcn_tensor_load_to_lds(g0, g1, z4, z4, z8, 0);
}

// ---------------------------------------------------------------------------
// LayerNorm: one wave per row of 512; optionally also emits raw input as bf16
// ---------------------------------------------------------------------------
__global__ __launch_bounds__(256) void ln_rows_kernel(
    const float* __restrict__ X, const float* __restrict__ sc,
    const float* __restrict__ bi, bf16_t* __restrict__ Y,
    bf16_t* __restrict__ Raw)
{
    const int lane = threadIdx.x & 31;
    const int wave = threadIdx.x >> 5;
    const int row  = blockIdx.x * 8 + wave;
    const float* xr = X + (size_t)row * DD;

    float x[16];
    float sum = 0.f, sq = 0.f;
#pragma unroll
    for (int e = 0; e < 16; ++e) {
        x[e] = xr[e * 32 + lane];
        sum += x[e];
        sq  += x[e] * x[e];
    }
#pragma unroll
    for (int m = 1; m < 32; m <<= 1) {
        sum += __shfl_xor(sum, m, 32);
        sq  += __shfl_xor(sq,  m, 32);
    }
    const float mean = sum * (1.f / (float)DD);
    const float var  = sq  * (1.f / (float)DD) - mean * mean;
    const float r    = rsqrtf(var + 1e-12f);

#pragma unroll
    for (int e = 0; e < 16; ++e) {
        const int c = e * 32 + lane;
        Y[(size_t)row * DD + c] = (bf16_t)((x[e] - mean) * r * sc[c] + bi[c]);
        if (Raw) Raw[(size_t)row * DD + c] = (bf16_t)x[e];
    }
}

// ---------------------------------------------------------------------------
// Attention mask bias: 0 if key allowed, -1e30 otherwise (diag handled later)
// ---------------------------------------------------------------------------
__global__ void mask_bias_kernel(const int* __restrict__ tr,
                                 const int* __restrict__ ba,
                                 float* __restrict__ mb, int n)
{
    int t = blockIdx.x * 256 + threadIdx.x;
    if (t < n) mb[t] = ((tr[t] + ba[t]) != 0) ? 0.f : -1e30f;
}

// ---------------------------------------------------------------------------
// Weight pack: fp32 W[K][N]  ->  bf16 W^T[N][K]  (coalesced writes)
// ---------------------------------------------------------------------------
__global__ void wpack_kernel(const float* __restrict__ W,
                             bf16_t* __restrict__ Wt, int K, int N)
{
    int t = blockIdx.x * 256 + threadIdx.x;
    if (t < K * N) {
        int n = t / K, k = t - n * K;
        Wt[t] = (bf16_t)W[(size_t)k * N + n];
    }
}

// ---------------------------------------------------------------------------
// TDM-staged WMMA GEMM: C[M][N] = A[M][K] * Bt[N][K]^T + bias (+ residual)
// block = 256 thr = 8 waves; block tile 128x128; wave tile 32x64.
// k-chunks of 64 are DMAed into double-buffered LDS by wave 0 via the
// Tensor Data Mover, synced with s_wait_tensorcnt + workgroup barrier.
// All fragments of a k-step live simultaneously so ds_loads clause up and
// drain under the WMMAs instead of serializing on one register set.
// ---------------------------------------------------------------------------
__global__ __launch_bounds__(256) void gemm_tdm_kernel(
    const bf16_t* __restrict__ A, const bf16_t* __restrict__ Bt,
    const float* __restrict__ bias, const float* __restrict__ residual,
    float* __restrict__ Cf, bf16_t* __restrict__ Cb, bf16_t* __restrict__ Ct,
    int M, int N, int K)
{
    __shared__ __align__(16) bf16_t Abuf[2][128][72];
    __shared__ __align__(16) bf16_t Bbuf[2][128][72];

    const int lane  = threadIdx.x & 31;
    const int wave  = threadIdx.x >> 5;
    const int l16   = lane & 15;
    const int half  = lane >> 4;
    const int waveM = wave & 3;
    const int waveN = wave >> 2;
    const int mBase = blockIdx.x * 128 + waveM * 32;
    const int nBase = blockIdx.y * 128 + waveN * 64;

    v8f acc[2][4];
#pragma unroll
    for (int s = 0; s < 2; ++s)
#pragma unroll
        for (int t = 0; t < 4; ++t) {
            const float bv = bias ? bias[nBase + t * 16 + l16] : 0.0f;
#pragma unroll
            for (int i = 0; i < 8; ++i) acc[s][t][i] = bv;
        }

    const bf16_t* Ablk = A  + (size_t)blockIdx.x * 128 * K;
    const bf16_t* Bblk = Bt + (size_t)blockIdx.y * 128 * K;
    const int KT = K >> 6;

    if (wave == 0) {
        tdm_load_2d((unsigned int)(uintptr_t)&Abuf[0][0][0], Ablk, 128, K);
        tdm_load_2d((unsigned int)(uintptr_t)&Bbuf[0][0][0], Bblk, 128, K);
    }

    for (int kc = 0; kc < KT; ++kc) {
        const int buf = kc & 1;
        if (wave == 0) {
            if (kc + 1 < KT) {
                tdm_load_2d((unsigned int)(uintptr_t)&Abuf[buf ^ 1][0][0],
                            Ablk + (kc + 1) * 64, 128, K);
                tdm_load_2d((unsigned int)(uintptr_t)&Bbuf[buf ^ 1][0][0],
                            Bblk + (kc + 1) * 64, 128, K);
                __builtin_amdgcn_s_wait_tensorcnt(2);   // current chunk landed
            } else {
                __builtin_amdgcn_s_wait_tensorcnt(0);
            }
        }
        __syncthreads();

#pragma unroll
        for (int ks = 0; ks < 2; ++ks) {
            // All 2 A-fragments + 4 B-fragments of this k-step are loaded into
            // distinct registers first (one big ds clause), then 8 WMMAs.
            FragBF a0, a1, b[4];
            const bf16_t* ap0 = &Abuf[buf][waveM * 32 + l16][ks * 32 + half * 8];
            a0.u[0] = *reinterpret_cast<const uint4*>(ap0);
            a0.u[1] = *reinterpret_cast<const uint4*>(ap0 + 16);
            const bf16_t* ap1 = ap0 + 16 * 72;
            a1.u[0] = *reinterpret_cast<const uint4*>(ap1);
            a1.u[1] = *reinterpret_cast<const uint4*>(ap1 + 16);
#pragma unroll
            for (int t = 0; t < 4; ++t) {
                const bf16_t* bp = &Bbuf[buf][waveN * 64 + t * 16 + l16][ks * 32 + half * 8];
                b[t].u[0] = *reinterpret_cast<const uint4*>(bp);
                b[t].u[1] = *reinterpret_cast<const uint4*>(bp + 16);
            }
#pragma unroll
            for (int t = 0; t < 4; ++t) {
                acc[0][t] = wmma_bf16(a0.v, b[t].v, acc[0][t]);
                acc[1][t] = wmma_bf16(a1.v, b[t].v, acc[1][t]);
            }
        }
        __syncthreads();
    }

#pragma unroll
    for (int s = 0; s < 2; ++s)
#pragma unroll
        for (int t = 0; t < 4; ++t) {
            const int col = nBase + t * 16 + l16;
#pragma unroll
            for (int i = 0; i < 8; ++i) {
                const int row = mBase + s * 16 + half * 8 + i;
                const size_t idx = (size_t)row * N + col;
                float val = acc[s][t][i];
                if (residual) val += residual[idx];
                if (Cf) Cf[idx] = val;
                if (Cb) Cb[idx] = (bf16_t)val;
                if (Ct) Ct[(size_t)col * M + row] = (bf16_t)val;
            }
        }
}

// ---------------------------------------------------------------------------
// TDM-staged gated FFN GEMM: out = bf16( gelu_new(A@W0) * (A@W1) )
// block tile 128x64; wave tile 32x32; A + two B tiles staged by TDM.
// ---------------------------------------------------------------------------
__global__ __launch_bounds__(256) void gemm_gated_tdm_kernel(
    const bf16_t* __restrict__ A, const bf16_t* __restrict__ B0t,
    const bf16_t* __restrict__ B1t, bf16_t* __restrict__ Out,
    int M, int N, int K)
{
    __shared__ __align__(16) bf16_t Abuf[2][128][72];
    __shared__ __align__(16) bf16_t B0buf[2][64][72];
    __shared__ __align__(16) bf16_t B1buf[2][64][72];

    const int lane  = threadIdx.x & 31;
    const int wave  = threadIdx.x >> 5;
    const int l16   = lane & 15;
    const int half  = lane >> 4;
    const int waveM = wave & 3;
    const int waveN = wave >> 2;
    const int mBase = blockIdx.x * 128 + waveM * 32;
    const int nBase = blockIdx.y * 64 + waveN * 32;

    v8f acc0[2][2], acc1[2][2];
#pragma unroll
    for (int s = 0; s < 2; ++s)
#pragma unroll
        for (int t = 0; t < 2; ++t)
#pragma unroll
            for (int i = 0; i < 8; ++i) { acc0[s][t][i] = 0.f; acc1[s][t][i] = 0.f; }

    const bf16_t* Ablk  = A   + (size_t)blockIdx.x * 128 * K;
    const bf16_t* B0blk = B0t + (size_t)blockIdx.y * 64 * K;
    const bf16_t* B1blk = B1t + (size_t)blockIdx.y * 64 * K;
    const int KT = K >> 6;

    if (wave == 0) {
        tdm_load_2d((unsigned int)(uintptr_t)&Abuf[0][0][0],  Ablk,  128, K);
        tdm_load_2d((unsigned int)(uintptr_t)&B0buf[0][0][0], B0blk, 64,  K);
        tdm_load_2d((unsigned int)(uintptr_t)&B1buf[0][0][0], B1blk, 64,  K);
    }

    for (int kc = 0; kc < KT; ++kc) {
        const int buf = kc & 1;
        if (wave == 0) {
            if (kc + 1 < KT) {
                tdm_load_2d((unsigned int)(uintptr_t)&Abuf[buf ^ 1][0][0],
                            Ablk + (kc + 1) * 64, 128, K);
                tdm_load_2d((unsigned int)(uintptr_t)&B0buf[buf ^ 1][0][0],
                            B0blk + (kc + 1) * 64, 64, K);
                tdm_load_2d((unsigned int)(uintptr_t)&B1buf[buf ^ 1][0][0],
                            B1blk + (kc + 1) * 64, 64, K);
                __builtin_amdgcn_s_wait_tensorcnt(3);
            } else {
                __builtin_amdgcn_s_wait_tensorcnt(0);
            }
        }
        __syncthreads();

#pragma unroll
        for (int ks = 0; ks < 2; ++ks) {
            FragBF a0, a1, b0[2], b1[2];
            const bf16_t* ap0 = &Abuf[buf][waveM * 32 + l16][ks * 32 + half * 8];
            a0.u[0] = *reinterpret_cast<const uint4*>(ap0);
            a0.u[1] = *reinterpret_cast<const uint4*>(ap0 + 16);
            const bf16_t* ap1 = ap0 + 16 * 72;
            a1.u[0] = *reinterpret_cast<const uint4*>(ap1);
            a1.u[1] = *reinterpret_cast<const uint4*>(ap1 + 16);
#pragma unroll
            for (int t = 0; t < 2; ++t) {
                const bf16_t* bp0 = &B0buf[buf][waveN * 32 + t * 16 + l16][ks * 32 + half * 8];
                b0[t].u[0] = *reinterpret_cast<const uint4*>(bp0);
                b0[t].u[1] = *reinterpret_cast<const uint4*>(bp0 + 16);
                const bf16_t* bp1 = &B1buf[buf][waveN * 32 + t * 16 + l16][ks * 32 + half * 8];
                b1[t].u[0] = *reinterpret_cast<const uint4*>(bp1);
                b1[t].u[1] = *reinterpret_cast<const uint4*>(bp1 + 16);
            }
#pragma unroll
            for (int t = 0; t < 2; ++t) {
                acc0[0][t] = wmma_bf16(a0.v, b0[t].v, acc0[0][t]);
                acc0[1][t] = wmma_bf16(a1.v, b0[t].v, acc0[1][t]);
                acc1[0][t] = wmma_bf16(a0.v, b1[t].v, acc1[0][t]);
                acc1[1][t] = wmma_bf16(a1.v, b1[t].v, acc1[1][t]);
            }
        }
        __syncthreads();
    }

#pragma unroll
    for (int s = 0; s < 2; ++s)
#pragma unroll
        for (int t = 0; t < 2; ++t) {
            const int col = nBase + t * 16 + l16;
#pragma unroll
            for (int i = 0; i < 8; ++i) {
                const int row = mBase + s * 16 + half * 8 + i;
                Out[(size_t)row * N + col] =
                    (bf16_t)(gelu_new(acc0[s][t][i]) * acc1[s][t][i]);
            }
        }
}

// ---------------------------------------------------------------------------
// Flash-style attention. grid = (N/64, B*H), block = 128 (4 waves).
// Each wave: 16 queries x full key stream (32 keys/iter) of one (b, head).
// Q,K row-major bf16 [8192][512]; V transposed bf16 [512][8192].
// ---------------------------------------------------------------------------
__global__ __launch_bounds__(128) void attn_kernel(
    const bf16_t* __restrict__ Qb, const bf16_t* __restrict__ Kb,
    const bf16_t* __restrict__ Vt, const float* __restrict__ maskb,
    bf16_t* __restrict__ Ob)
{
    __shared__ __align__(16) bf16_t plds[4][16][40];   // padded: conflict-free

    const int lane = threadIdx.x & 31;
    const int wave = threadIdx.x >> 5;
    const int l16  = lane & 15;
    const int half = lane >> 4;
    const int bh   = blockIdx.y;
    const int b    = bh >> 3;
    const int h0   = bh & 7;
    const int qBase = blockIdx.x * 64 + wave * 16;
    const float scale = 0.044194173824159216f;         // 1/sqrt(D=512)

    FragBF qf0, qf1;
    {
        const bf16_t* qp = Qb + (size_t)(b * NN + qBase + l16) * DD + h0 * DHH + half * 8;
        qf0.u[0] = *reinterpret_cast<const uint4*>(qp);
        qf0.u[1] = *reinterpret_cast<const uint4*>(qp + 16);
        qf1.u[0] = *reinterpret_cast<const uint4*>(qp + 32);
        qf1.u[1] = *reinterpret_cast<const uint4*>(qp + 48);
    }

    v8f acc[4];
#pragma unroll
    for (int t = 0; t < 4; ++t)
#pragma unroll
        for (int i = 0; i < 8; ++i) acc[t][i] = 0.f;
    float mrow[8], lrow[8];
#pragma unroll
    for (int i = 0; i < 8; ++i) { mrow[i] = -1e30f; lrow[i] = 0.f; }

    for (int jb = 0; jb < NN; jb += 32) {
        // --- scores for two 16-key tiles: distinct regs for all 4 K-frags --
        v8f sA, sB;
        {
            FragBF k0a, k1a, k0b, k1b;
            const bf16_t* kpA = Kb + (size_t)(b * NN + jb + l16) * DD + h0 * DHH + half * 8;
            k0a.u[0] = *reinterpret_cast<const uint4*>(kpA);
            k0a.u[1] = *reinterpret_cast<const uint4*>(kpA + 16);
            k1a.u[0] = *reinterpret_cast<const uint4*>(kpA + 32);
            k1a.u[1] = *reinterpret_cast<const uint4*>(kpA + 48);
            const bf16_t* kpB = kpA + (size_t)16 * DD;
            k0b.u[0] = *reinterpret_cast<const uint4*>(kpB);
            k0b.u[1] = *reinterpret_cast<const uint4*>(kpB + 16);
            k1b.u[0] = *reinterpret_cast<const uint4*>(kpB + 32);
            k1b.u[1] = *reinterpret_cast<const uint4*>(kpB + 48);
            v8f z;
#pragma unroll
            for (int i = 0; i < 8; ++i) z[i] = 0.f;
            sA = wmma_bf16(qf0.v, k0a.v, z);
            sB = wmma_bf16(qf0.v, k0b.v, z);
            sA = wmma_bf16(qf1.v, k1a.v, sA);
            sB = wmma_bf16(qf1.v, k1b.v, sB);
        }

        const int   keyA  = jb + l16;
        const int   keyB  = jb + 16 + l16;
        const float biasA = maskb[b * NN + keyA];
        const float biasB = maskb[b * NN + keyB];
        float pA[8], pB[8], alpha[8];
#pragma unroll
        for (int i = 0; i < 8; ++i) {
            const int q = qBase + half * 8 + i;
            float va = sA[i] * scale; va = (keyA == q) ? va : va + biasA;
            float vb = sB[i] * scale; vb = (keyB == q) ? vb : vb + biasB;
            float mloc = fmaxf(va, vb);
#pragma unroll
            for (int m = 1; m < 16; m <<= 1) mloc = fmaxf(mloc, __shfl_xor(mloc, m, 32));
            const float mnew = fmaxf(mrow[i], mloc);
            alpha[i] = __expf(mrow[i] - mnew);
            va = __expf(va - mnew);
            vb = __expf(vb - mnew);
            float rs = va + vb;
#pragma unroll
            for (int m = 1; m < 16; m <<= 1) rs += __shfl_xor(rs, m, 32);
            lrow[i] = lrow[i] * alpha[i] + rs;
            mrow[i] = mnew;
            pA[i] = va; pB[i] = vb;
        }

        // --- reshape P (C-layout) -> A-fragment via LDS --------------------
#pragma unroll
        for (int i = 0; i < 8; ++i) {
            plds[wave][half * 8 + i][l16]      = (bf16_t)pA[i];
            plds[wave][half * 8 + i][16 + l16] = (bf16_t)pB[i];
        }
        __syncthreads();
        FragBF pf;
        {
            const bf16_t* pp = &plds[wave][l16][half * 8];
            pf.u[0] = *reinterpret_cast<const uint4*>(pp);
            pf.u[1] = *reinterpret_cast<const uint4*>(pp + 16);
        }
        __syncthreads();

        // --- O += P * V: all 4 V-frags live at once, then 4 WMMAs ----------
        FragBF vf[4];
#pragma unroll
        for (int t = 0; t < 4; ++t) {
            const bf16_t* vp = Vt + (size_t)(h0 * DHH + t * 16 + l16) * MM
                                  + b * NN + jb + half * 8;
            vf[t].u[0] = *reinterpret_cast<const uint4*>(vp);
            vf[t].u[1] = *reinterpret_cast<const uint4*>(vp + 16);
        }
#pragma unroll
        for (int t = 0; t < 4; ++t) {
#pragma unroll
            for (int i = 0; i < 8; ++i) acc[t][i] *= alpha[i];
            acc[t] = wmma_bf16(pf.v, vf[t].v, acc[t]);
        }
    }

#pragma unroll
    for (int t = 0; t < 4; ++t)
#pragma unroll
        for (int i = 0; i < 8; ++i) {
            const int q = qBase + half * 8 + i;
            Ob[(size_t)(b * NN + q) * DD + h0 * DHH + t * 16 + l16] =
                (bf16_t)(acc[t][i] / lrow[i]);
        }
}

// ---------------------------------------------------------------------------
// Host orchestration
// ---------------------------------------------------------------------------
extern "C" void kernel_launch(void* const* d_in, const int* in_sizes, int n_in,
                              void* d_out, int out_size, void* d_ws, size_t ws_size,
                              hipStream_t stream)
{
    (void)in_sizes; (void)n_in; (void)out_size; (void)ws_size;

    const float* X      = (const float*)d_in[0];
    const int*   tr     = (const int*)  d_in[1];
    const int*   ba     = (const int*)  d_in[2];
    const float* Wq     = (const float*)d_in[3];
    const float* bq     = (const float*)d_in[4];
    const float* Wk     = (const float*)d_in[5];
    const float* bk     = (const float*)d_in[6];
    const float* Wv     = (const float*)d_in[7];
    const float* bv     = (const float*)d_in[8];
    const float* Wmix   = (const float*)d_in[9];
    const float* bmix   = (const float*)d_in[10];
    const float* ln0_s  = (const float*)d_in[11];
    const float* ln0_b  = (const float*)d_in[12];
    const float* ln1_s  = (const float*)d_in[13];
    const float* ln1_b  = (const float*)d_in[14];
    const float* wi0    = (const float*)d_in[15];
    const float* wi1    = (const float*)d_in[16];
    const float* wo     = (const float*)d_in[17];
    float* out = (float*)d_out;

    char* w = (char*)d_ws;
    size_t cur = 0;
    auto take = [&](size_t bytes) -> char* {
        char* p = w + cur;
        cur += (bytes + 255) & ~(size_t)255;
        return p;
    };

    bf16_t* Xbf   = (bf16_t*)take((size_t)MM * DD * 2);   // raw X, bf16
    bf16_t* Xnbf  = (bf16_t*)take((size_t)MM * DD * 2);   // LN0(X); reused as x1
    float*  Qf32  = (float*) take((size_t)MM * DD * 4);   // Q (fp32 residual)
    bf16_t* Qbf   = (bf16_t*)take((size_t)MM * DD * 2);
    bf16_t* Kbf   = (bf16_t*)take((size_t)MM * DD * 2);
    bf16_t* Vtbf  = (bf16_t*)take((size_t)DD * MM * 2);   // V^T [512][8192]
    bf16_t* Obf   = (bf16_t*)take((size_t)MM * DD * 2);
    float*  Hc    = (float*) take((size_t)MM * DD * 4);
    float*  maskb = (float*) take((size_t)BB * NN * 4);
    bf16_t* Wqt   = (bf16_t*)take((size_t)DD * DD * 2);
    bf16_t* Wkt   = (bf16_t*)take((size_t)DD * DD * 2);
    bf16_t* Wvt   = (bf16_t*)take((size_t)DD * DD * 2);
    bf16_t* Wmixt = (bf16_t*)take((size_t)DD * DD * 2);
    bf16_t* wi0t  = (bf16_t*)take((size_t)DD * FFD * 2);
    bf16_t* wi1t  = (bf16_t*)take((size_t)DD * FFD * 2);
    bf16_t* wot   = (bf16_t*)take((size_t)FFD * DD * 2);
    // FFN activation (32 MB) aliases the dead Qf32+Qbf+Kbf region (16+8+8 MB)
    bf16_t* ffin  = (bf16_t*)Qf32;
    bf16_t* x1bf  = Xnbf;

    // 1) LN0 (+ raw X -> bf16)
    ln_rows_kernel<<<MM / 8, 256, 0, stream>>>(X, ln0_s, ln0_b, Xnbf, Xbf);

    // 2) mask bias
    mask_bias_kernel<<<(BB * NN + 255) / 256, 256, 0, stream>>>(tr, ba, maskb, BB * NN);

    // 3) pack weights -> bf16 W^T
    wpack_kernel<<<(DD * DD)  / 256, 256, 0, stream>>>(Wq,   Wqt,   DD,  DD);
    wpack_kernel<<<(DD * DD)  / 256, 256, 0, stream>>>(Wk,   Wkt,   DD,  DD);
    wpack_kernel<<<(DD * DD)  / 256, 256, 0, stream>>>(Wv,   Wvt,   DD,  DD);
    wpack_kernel<<<(DD * DD)  / 256, 256, 0, stream>>>(Wmix, Wmixt, DD,  DD);
    wpack_kernel<<<(DD * FFD) / 256, 256, 0, stream>>>(wi0,  wi0t,  DD,  FFD);
    wpack_kernel<<<(DD * FFD) / 256, 256, 0, stream>>>(wi1,  wi1t,  DD,  FFD);
    wpack_kernel<<<(FFD * DD) / 256, 256, 0, stream>>>(wo,   wot,   FFD, DD);

    // 4) Q/K/V projections (WMMA, TDM-staged)
    dim3 g512(MM / 128, DD / 128);
    gemm_tdm_kernel<<<g512, 256, 0, stream>>>(Xnbf, Wqt, bq, nullptr,
                                              Qf32, Qbf, nullptr, MM, DD, DD);
    gemm_tdm_kernel<<<g512, 256, 0, stream>>>(Xbf, Wkt, bk, nullptr,
                                              nullptr, Kbf, nullptr, MM, DD, DD);
    gemm_tdm_kernel<<<g512, 256, 0, stream>>>(Xbf, Wvt, bv, nullptr,
                                              nullptr, nullptr, Vtbf, MM, DD, DD);

    // 5) attention
    attn_kernel<<<dim3(NN / 64, BB * HH), 128, 0, stream>>>(Qbf, Kbf, Vtbf, maskb, Obf);

    // 6) head mix + residual (Q)
    gemm_tdm_kernel<<<g512, 256, 0, stream>>>(Obf, Wmixt, bmix, Qf32,
                                              Hc, nullptr, nullptr, MM, DD, DD);

    // 7) LN1
    ln_rows_kernel<<<MM / 8, 256, 0, stream>>>(Hc, ln1_s, ln1_b, x1bf, nullptr);

    // 8) gated GEGLU FFN up-projection (TDM-staged)
    gemm_gated_tdm_kernel<<<dim3(MM / 128, FFD / 64), 256, 0, stream>>>(
        x1bf, wi0t, wi1t, ffin, MM, FFD, DD);

    // 9) FFN down-projection + residual (Hc) -> fp32 output
    gemm_tdm_kernel<<<g512, 256, 0, stream>>>(ffin, wot, nullptr, Hc,
                                              out, nullptr, nullptr, MM, DD, FFD);
}
